// AFTLocalAttention_13597866459775
// MI455X (gfx1250) — matrix operand order repaired and checked by reading, not compiled, not run on hardware
//
#include <hip/hip_runtime.h>
#include <hip/hip_bf16.h>

#define D_MODEL 512
#define SEQ     2048
#define WIN     128
#define BATCH   8
#define MTOT    (BATCH*SEQ)          /* 16384 */
#define BSD     (BATCH*SEQ*D_MODEL)  /* 8388608 */
#define NKB     9                    /* banded k-steps of 32 */
#define BANDK   (NKB*32)             /* 288 */
#define PAD     160                  /* t-halo so the band sweep never clamps */
#define TPAD    (PAD + SEQ + PAD)    /* 2368, padded t-stride */

typedef _Float16 h16;
typedef __attribute__((ext_vector_type(16))) _Float16 v16h;
typedef __attribute__((ext_vector_type(8)))  _Float16 v8h;
typedef __attribute__((ext_vector_type(4)))  _Float16 v4h;
typedef __attribute__((ext_vector_type(8)))  float    v8f;
typedef __attribute__((ext_vector_type(4)))  float    v4f;

// ---- WMMA operand loaders (pure f16, all 16B-aligned b128 loads) ----------
// A (16x32 f16): lane l holds row m=l&15; halves 0-15 = 16 consecutive k
// starting at (l>>4)*16 (klo passed as kt + (l>>4)*8).
static __device__ inline v16h load_a16_h(const h16* row, int klo) {
  v8h lo = *(const v8h*)(row + klo + (klo & 15));
  v8h hi = *(const v8h*)(row + klo + (klo & 15) + 8);
  return __builtin_shufflevector(lo, hi, 0, 1, 2, 3, 4, 5, 6, 7,
                                 8, 9, 10, 11, 12, 13, 14, 15);
}
// B (32x16 f16): lane l holds column n=l&15; halves 0-15 = 16 consecutive k
// starting at (l>>4)*16.
static __device__ inline v16h load_b16_h(const h16* p) {
  v8h lo = *(const v8h*)(p);
  v8h hi = *(const v8h*)(p + 8);
  return __builtin_shufflevector(lo, hi, 0, 1, 2, 3, 4, 5, 6, 7,
                                 8, 9, 10, 11, 12, 13, 14, 15);
}

#define WMMA_F16(A, B, C) \
  __builtin_amdgcn_wmma_f32_16x16x32_f16(false, (A), false, (B), (short)0, (C), false, false)

// ---- Kernel 0a: bulk f32 -> f16 conversion (vectorized) -------------------
__global__ void aft_cvt_kernel(const float* __restrict__ src,
                               h16* __restrict__ dst, int n4) {
  const int i = blockIdx.x * blockDim.x + threadIdx.x;
  if (i >= n4) return;
  v4f x = ((const v4f*)src)[i];
  v4h y;
#pragma unroll
  for (int j = 0; j < 4; ++j) y[j] = (h16)x[j];
  ((v4h*)dst)[i] = y;
}

// ---- Kernel 0b: zero the t-halo of the padded ekT/ekvT buffers ------------
__global__ void aft_padzero_kernel(h16* __restrict__ ekT, h16* __restrict__ ekvT) {
  const int idx = blockIdx.x * blockDim.x + threadIdx.x;
  if (idx >= BATCH * D_MODEL * 2 * PAD) return;
  const int row = idx / (2 * PAD);
  const int j   = idx % (2 * PAD);
  const int off = (j < PAD) ? j : (SEQ + j);     // left halo | right halo
  const size_t p = (size_t)row * TPAD + off;
  ekT[p]  = (h16)0.0f;
  ekvT[p] = (h16)0.0f;
}

// ---- Kernel 1: fused Q/K/V projections + activations ----------------------
// Each wave: one 16x32 output tile (2 n-sub-tiles share each A operand).
// Inner loop is pure b128 f16 loads + 6 WMMAs per k-step.
__global__ __launch_bounds__(128)
void aft_proj_kernel(const h16* __restrict__ q_h, const h16* __restrict__ k_h,
                     const h16* __restrict__ v_h,
                     const h16* __restrict__ wq_h, const float* __restrict__ bq,
                     const h16* __restrict__ wk_h, const float* __restrict__ bk,
                     const h16* __restrict__ wv_h, const float* __restrict__ bv,
                     float* __restrict__ qsig, h16* __restrict__ ekT,
                     h16* __restrict__ ekvT) {
  const int lane = threadIdx.x & 31;
  const int wave = threadIdx.x >> 5;
  const int tile = blockIdx.x * 4 + wave;        // (MTOT/16)*(D/32) = 16384
  const int nt = tile & 15, mt = tile >> 4;
  const int m0 = mt << 4, n0 = nt << 5;

  const int ar = m0 + (lane & 15);
  const int kh = (lane >> 4) * 8;
  const int kb = (lane >> 4) * 16;

  const h16* xin[3] = {q_h + (size_t)ar * D_MODEL, k_h + (size_t)ar * D_MODEL,
                       v_h + (size_t)ar * D_MODEL};
  const h16* wr[3][2];
  const h16* Wm[3] = {wq_h, wk_h, wv_h};
#pragma unroll
  for (int g = 0; g < 3; ++g) {
#pragma unroll
    for (int n = 0; n < 2; ++n)
      wr[g][n] = Wm[g] + (size_t)(n0 + 16 * n + (lane & 15)) * D_MODEL;
  }

  v8f acc[3][2] = {};
  for (int kt = 0; kt < D_MODEL; kt += 32) {
#pragma unroll
    for (int g = 0; g < 3; ++g) {
      v16h a = load_a16_h(xin[g], kt + kh);
#pragma unroll
      for (int n = 0; n < 2; ++n) {
        v16h b = load_b16_h(wr[g][n] + kt + kb);
        acc[g][n] = WMMA_F16(a, b, acc[g][n]);
      }
    }
  }

#pragma unroll
  for (int n = 0; n < 2; ++n) {
    const int col = n0 + 16 * n + (lane & 15);
    const float biasq = bq[col], biask = bk[col], biasv = bv[col];
#pragma unroll
    for (int i = 0; i < 8; ++i) {
      const int mg = m0 + i + ((lane >> 4) << 3);  // C row = i + 8*(lane>=16)
      const float xq = acc[0][n][i] + biasq;
      const float xk = acc[1][n][i] + biask;
      const float xv = acc[2][n][i] + biasv;
      const float sg  = 1.0f / (1.0f + __expf(-xq));
      const float ek  = __expf(xk);
      const float ekv = ek * xv;
      qsig[(size_t)mg * D_MODEL + col] = sg;
      const int b = mg >> 11;          // / SEQ
      const int t = mg & (SEQ - 1);
      const size_t tp = ((size_t)b * D_MODEL + col) * TPAD + PAD + t;
      ekT[tp]  = (h16)ek;
      ekvT[tp] = (h16)ekv;
    }
  }
}

// ---- Kernel 2: column sums over t ("background 1" term) -------------------
__global__ void aft_colsum_kernel(const h16* __restrict__ ekT,
                                  const h16* __restrict__ ekvT,
                                  float* __restrict__ cs_ek,
                                  float* __restrict__ cs_ekv) {
  const int idx = blockIdx.x * blockDim.x + threadIdx.x;  // b*D + d
  if (idx >= BATCH * D_MODEL) return;
  const h16* p  = ekT  + (size_t)idx * TPAD + PAD;
  const h16* pv = ekvT + (size_t)idx * TPAD + PAD;
  float s = 0.f, sv = 0.f;
  for (int t = 0; t < SEQ; ++t) { s += (float)p[t]; sv += (float)pv[t]; }
  cs_ek[idx]  = s;
  cs_ekv[idx] = sv;
}

// ---- Kernel 3: packed band of exp(bias)-1, f16, aligned to WMMA sweep -----
__global__ void aft_ewm1_kernel(const float* __restrict__ wb,
                                h16* __restrict__ ewm1) {
  const int idx = blockIdx.x * blockDim.x + threadIdx.x;
  if (idx >= SEQ * BANDK) return;
  const int j = idx % BANDK;
  const int s = idx / BANDK;
  const int s0 = s & ~15;
  const int tstart = ((s0 >> 5) << 5) - 128;  // 9 k-steps cover [s0-127, s0+142]
  const int t = tstart + j;
  int d = s - t; if (d < 0) d = -d;
  float val = 0.0f;
  if (t >= 0 && t < SEQ && d < WIN)
    val = __expf(wb[(size_t)s * SEQ + t]) - 1.0f;
  ewm1[(size_t)s * BANDK + j] = (h16)val;
}

// ---- Kernel 4: banded num/den WMMA + combine ------------------------------
// Each wave: 16 rows x 32 d-columns; one A load feeds 4 WMMAs (num/den x 2).
__global__ __launch_bounds__(128)
void aft_band_kernel(const h16* __restrict__ ewm1, const h16* __restrict__ ekT,
                     const h16* __restrict__ ekvT,
                     const float* __restrict__ cs_ek,
                     const float* __restrict__ cs_ekv,
                     const float* __restrict__ qsig,
                     float* __restrict__ out_num, h16* __restrict__ x_h) {
  const int lane = threadIdx.x & 31;
  const int wave = threadIdx.x >> 5;
  const int tile = blockIdx.x * 4 + wave;      // B*(S/16)*(D/32) = 16384
  const int dt = tile & 15;
  const int st = (tile >> 4) & 127;
  const int b  = tile >> 11;
  const int s0 = st << 4, d0 = dt << 5;
  const int tstart = ((s0 >> 5) << 5) - 128;   // >= -144, halo covers it

  const int srow = s0 + (lane & 15);
  const int kh   = (lane >> 4) * 8;
  const int kb   = (lane >> 4) * 16;

  const h16* arow = ewm1 + (size_t)srow * BANDK;
  const h16* ekrow[2];
  const h16* ekvrow[2];
#pragma unroll
  for (int n = 0; n < 2; ++n) {
    const size_t r = ((size_t)b * D_MODEL + d0 + 16 * n + (lane & 15)) * TPAD;
    ekrow[n]  = ekT  + r + PAD + tstart;    // in-bounds thanks to halo
    ekvrow[n] = ekvT + r + PAD + tstart;
  }

  v8f cn[2] = {}; v8f cd[2] = {};
  for (int kt = 0; kt < BANDK; kt += 32) {
    v16h a = load_a16_h(arow, kt + kh);
#pragma unroll
    for (int n = 0; n < 2; ++n) {
      v16h bn = load_b16_h(ekvrow[n] + kt + kb);
      cn[n] = WMMA_F16(a, bn, cn[n]);
      v16h bd = load_b16_h(ekrow[n] + kt + kb);
      cd[n] = WMMA_F16(a, bd, cd[n]);
    }
  }

#pragma unroll
  for (int n = 0; n < 2; ++n) {
    const int dcol = d0 + 16 * n + (lane & 15);
    const float csk  = cs_ek[b * D_MODEL + dcol];
    const float cskv = cs_ekv[b * D_MODEL + dcol];
#pragma unroll
    for (int i = 0; i < 8; ++i) {
      const int s = s0 + i + ((lane >> 4) << 3);
      const size_t idx = ((size_t)b * SEQ + s) * D_MODEL + dcol;
      const float numv = cskv + cn[n][i];
      const float denv = csk + cd[n][i];
      const float numq = qsig[idx] * numv;
      out_num[idx] = numq;                // second tuple output
      x_h[idx] = (h16)(numq / denv);      // f16 operand for final GEMM
    }
  }
}

// ---- Kernel 5: output projection ------------------------------------------
__global__ __launch_bounds__(128)
void aft_outgemm_kernel(const h16* __restrict__ x_h, const h16* __restrict__ wo_h,
                        const float* __restrict__ bo, float* __restrict__ out) {
  const int lane = threadIdx.x & 31;
  const int wave = threadIdx.x >> 5;
  const int tile = blockIdx.x * 4 + wave;      // (MTOT/16)*(D/32) = 16384
  const int nt = tile & 15, mt = tile >> 4;
  const int m0 = mt << 4, n0 = nt << 5;

  const int ar = m0 + (lane & 15);
  const int kh = (lane >> 4) * 8;
  const int kb = (lane >> 4) * 16;

  const h16* xrow = x_h + (size_t)ar * D_MODEL;
  const h16* wor[2];
#pragma unroll
  for (int n = 0; n < 2; ++n)
    wor[n] = wo_h + (size_t)(n0 + 16 * n + (lane & 15)) * D_MODEL;

  v8f c[2] = {};
  for (int kt = 0; kt < D_MODEL; kt += 32) {
    v16h a = load_a16_h(xrow, kt + kh);
#pragma unroll
    for (int n = 0; n < 2; ++n) {
      v16h b = load_b16_h(wor[n] + kt + kb);
      c[n] = WMMA_F16(a, b, c[n]);
    }
  }
#pragma unroll
  for (int n = 0; n < 2; ++n) {
    const int col = n0 + 16 * n + (lane & 15);
    const float bias = bo[col];
#pragma unroll
    for (int i = 0; i < 8; ++i) {
      const int mg = m0 + i + ((lane >> 4) << 3);
      out[(size_t)mg * D_MODEL + col] = c[n][i] + bias;
    }
  }
}

// ---- Launcher --------------------------------------------------------------
extern "C" void kernel_launch(void* const* d_in, const int* in_sizes, int n_in,
                              void* d_out, int out_size, void* d_ws, size_t ws_size,
                              hipStream_t stream) {
  const float* q     = (const float*)d_in[0];
  const float* k     = (const float*)d_in[1];
  const float* v     = (const float*)d_in[2];
  const float* Wq_w  = (const float*)d_in[3];
  const float* Wq_b  = (const float*)d_in[4];
  const float* Wk_w  = (const float*)d_in[5];
  const float* Wk_b  = (const float*)d_in[6];
  const float* Wv_w  = (const float*)d_in[7];
  const float* Wv_b  = (const float*)d_in[8];
  const float* wbias = (const float*)d_in[9];
  const float* out_w = (const float*)d_in[10];
  const float* out_b = (const float*)d_in[11];
  // d_in[12] = local_mask: band is computed analytically, not needed.

  char* ws = (char*)d_ws;
  float* qsig   = (float*)(ws);                              // 33,554,432 B
  h16*   ekT    = (h16*)  (ws + 33554432);                   // 19,398,656 B (padded)
  h16*   ekvT   = (h16*)  (ws + 52953088);                   // 19,398,656 B (padded)
  h16*   x_h    = (h16*)  (ws + 72351744);                   // 16,777,216 B
  float* cs_ek  = (float*)(ws + 89128960);                   //     16,384 B
  float* cs_ekv = (float*)(ws + 89145344);                   //     16,384 B
  h16*   ewm1   = (h16*)  (ws + 89161728);                   //  1,179,648 B
  h16*   q_h    = (h16*)  (ws + 90341376);                   // 16,777,216 B
  h16*   k_h    = (h16*)  (ws + 107118592);                  // 16,777,216 B
  h16*   v_h    = (h16*)  (ws + 123895808);                  // 16,777,216 B
  h16*   wq_h   = (h16*)  (ws + 140673024);                  //    524,288 B
  h16*   wk_h   = (h16*)  (ws + 141197312);                  //    524,288 B
  h16*   wv_h   = (h16*)  (ws + 141721600);                  //    524,288 B
  h16*   wo_h   = (h16*)  (ws + 142245888);                  //    524,288 B

  float* out     = (float*)d_out;
  float* out_num = out + (size_t)BSD;

  // One-shot f32 -> f16 conversions (vectorized, bandwidth-trivial).
  const int nbsd4 = BSD / 4;                     // 2,097,152
  const int nw4   = (D_MODEL * D_MODEL) / 4;     //    65,536
  aft_cvt_kernel<<<(nbsd4 + 255) / 256, 256, 0, stream>>>(q, q_h, nbsd4);
  aft_cvt_kernel<<<(nbsd4 + 255) / 256, 256, 0, stream>>>(k, k_h, nbsd4);
  aft_cvt_kernel<<<(nbsd4 + 255) / 256, 256, 0, stream>>>(v, v_h, nbsd4);
  aft_cvt_kernel<<<(nw4 + 255) / 256, 256, 0, stream>>>(Wq_w, wq_h, nw4);
  aft_cvt_kernel<<<(nw4 + 255) / 256, 256, 0, stream>>>(Wk_w, wk_h, nw4);
  aft_cvt_kernel<<<(nw4 + 255) / 256, 256, 0, stream>>>(Wv_w, wv_h, nw4);
  aft_cvt_kernel<<<(nw4 + 255) / 256, 256, 0, stream>>>(out_w, wo_h, nw4);

  aft_padzero_kernel<<<(BATCH * D_MODEL * 2 * PAD + 255) / 256, 256, 0, stream>>>(
      ekT, ekvT);
  // 16384 wave-tiles, 4 waves per 128-thread block -> 4096 blocks
  aft_proj_kernel<<<4096, 128, 0, stream>>>(q_h, k_h, v_h, wq_h, Wq_b, wk_h,
                                            Wk_b, wv_h, Wv_b, qsig, ekT, ekvT);
  aft_colsum_kernel<<<(BATCH * D_MODEL + 255) / 256, 256, 0, stream>>>(
      ekT, ekvT, cs_ek, cs_ekv);
  aft_ewm1_kernel<<<(SEQ * BANDK + 255) / 256, 256, 0, stream>>>(wbias, ewm1);
  aft_band_kernel<<<4096, 128, 0, stream>>>(ewm1, ekT, ekvT, cs_ek, cs_ekv,
                                            qsig, out_num, x_h);
  aft_outgemm_kernel<<<4096, 128, 0, stream>>>(x_h, out_w ? wo_h : wo_h, out_b, out);
}
